// FixedRevRNN_5385888989406
// MI455X (gfx1250) — matrix operand order repaired and checked by reading, not compiled, not run on hardware
//
#include <hip/hip_runtime.h>
#include <hip/hip_bf16.h>

typedef __attribute__((ext_vector_type(16))) __bf16 v16bf;
typedef __attribute__((ext_vector_type(8)))  __bf16 v8bf;
typedef __attribute__((ext_vector_type(8)))  float  v8f;
typedef __attribute__((ext_vector_type(4)))  float  v4f;

#define HF    256
#define FS    16
#define TSEQ  256
#define TTOT  264   // T + DELAY
#define BB    256
#define NDELAY 8

// ---------------- WMMA tile helpers (CDNA5 wave32 layouts) ----------------
// 16-bit A-matrix 16x32: lane<16 -> M=lane, k in {0..7,16..23}; lane>=16 -> M=lane-16, k in {8..15,24..31}.
// Row-major storage => per lane the fragment is two contiguous 16-byte chunks.
static __device__ inline v16bf loadA_bf16v(const __bf16* A, int lda, int k0, int lane) {
  int m = lane & 15, kb = (lane >> 4) * 8;
  const __bf16* rowp = A + (size_t)m * lda + k0 + kb;
  v8bf lo = *(const v8bf*)rowp;
  v8bf hi = *(const v8bf*)(rowp + 16);
  v16bf a;
#pragma unroll
  for (int i = 0; i < 8; ++i) { a[i] = lo[i]; a[8 + i] = hi[i]; }
  return a;
}

// f32 row-major source -> bf16 A fragment (4x 16-byte loads + cvt)
static __device__ inline v16bf loadA_f32v(const float* rowp0, int k0, int lane) {
  int kb = (lane >> 4) * 8;
  const float* p = rowp0 + k0 + kb;
  v4f x0 = *(const v4f*)p,        x1 = *(const v4f*)(p + 4);
  v4f x2 = *(const v4f*)(p + 16), x3 = *(const v4f*)(p + 20);
  v16bf a;
#pragma unroll
  for (int i = 0; i < 4; ++i) {
    a[i]      = (__bf16)x0[i];
    a[4 + i]  = (__bf16)x1[i];
    a[8 + i]  = (__bf16)x2[i];
    a[12 + i] = (__bf16)x3[i];
  }
  return a;
}

// B fragments pre-swizzled in workspace as [tile][lane][16] -> one 32-byte load per lane.
static __device__ inline v16bf loadB_swz(const __bf16* tileBase, int lane) {
  return *(const v16bf*)(tileBase + lane * 16);
}

static __device__ inline v8f wmma_bf16(v16bf a, v16bf b, v8f c) {
  return __builtin_amdgcn_wmma_f32_16x16x32_bf16(false, a, false, b, (short)0, c, false, false);
}

// ---------------- Kernel 1a: swizzle six 256x256 matrices into WMMA-B layout ----------------
// Wz element index = ((mat*8 + kc)*16 + nt)*512 + lane*16 + e ; value = M[k][col],
// k = kc*32 + (lane>>4)*8 + (e&7) + (e>>3)*16, col = nt*16 + (lane&15).
__global__ __launch_bounds__(256) void k_prepw256(const float* __restrict__ lp0,
                                                  const float* __restrict__ lp1,
                                                  __bf16* __restrict__ Wz) {
  int id   = blockIdx.x * 256 + threadIdx.x;     // 0..393215
  int e    = id & 15;
  int lane = (id >> 4) & 31;
  int nt   = (id >> 9) & 15;
  int kc   = (id >> 13) & 7;
  int mat  = id >> 16;                           // 0..5
  const float* base;
  switch (mat) {
    case 0:  base = lp0; break;                  // W0x  = lp0[0:256]
    case 1:  base = lp0 + 2 * HF * HF; break;    // W0o  = lp0[512:768]
    case 2:  base = lp1; break;                  // W1x
    case 3:  base = lp1 + 2 * HF * HF; break;    // W1o
    case 4:  base = lp0 + HF * HF; break;        // Wg0  = lp0[256:512]
    default: base = lp1 + HF * HF; break;        // Wg1
  }
  int k   = kc * 32 + (lane >> 4) * 8 + (e & 7) + (e >> 3) * 16;
  int col = nt * 16 + (lane & 15);
  Wz[id] = (__bf16)base[k * HF + col];
}

// ---------------- Kernel 1b: swizzle out_linear (512x1024) into WMMA-B layout ----------------
__global__ __launch_bounds__(256) void k_prepwout(const float* __restrict__ wout,
                                                  __bf16* __restrict__ Wz) {
  int id   = blockIdx.x * 256 + threadIdx.x;     // 0..524287
  int e    = id & 15;
  int lane = (id >> 4) & 31;
  int nt   = (id >> 9) & 63;
  int kc   = id >> 15;                           // 0..15
  int k    = kc * 32 + (lane >> 4) * 8 + (e & 7) + (e >> 3) * 16;
  int col  = nt * 16 + (lane & 15);
  Wz[id] = (__bf16)wout[(size_t)k * 1024 + col];
}

// ---------------- Kernel 2: gate precompute, stored in C-fragment layout ----------------
// G element index = (((t*16 + btile)*16 + nt)*32 + lane)*8 + i  (i = accumulator row)
__global__ __launch_bounds__(256) void k_gates(const int* __restrict__ fn_input,
                                               const float* __restrict__ emb,
                                               const __bf16* __restrict__ Wz,
                                               __bf16* __restrict__ G0,
                                               __bf16* __restrict__ G1) {
  int wave = threadIdx.x >> 5, lane = threadIdx.x & 31;
  int t  = blockIdx.x >> 4;
  int bt = blockIdx.x & 15;
  int b  = bt * 16 + (lane & 15);
  int idx = (t < TSEQ) ? fn_input[b * TSEQ + t] : 0;   // pad steps use embedding[0]
  const float* rowp = emb + (size_t)idx * HF;

  v8f acc[4] = {};                                     // 1 param x 4 n-tiles per wave
#pragma unroll 1
  for (int kc = 0; kc < 8; ++kc) {
    v16bf a = loadA_f32v(rowp, kc * 32, lane);
#pragma unroll
    for (int tt = 0; tt < 4; ++tt) {
      int tile = wave * 4 + tt;
      int p    = tile >> 4;
      int nt   = tile & 15;
      const __bf16* Bt = Wz + (size_t)(((4 + p) * 8 + kc) * 16 + nt) * 512;
      acc[tt] = wmma_bf16(a, loadB_swz(Bt, lane), acc[tt]);
    }
  }
#pragma unroll
  for (int tt = 0; tt < 4; ++tt) {
    int tile = wave * 4 + tt;
    int p    = tile >> 4;
    int nt   = tile & 15;
    __bf16* Gp = p ? G1 : G0;
    v8bf g;
#pragma unroll
    for (int i = 0; i < 8; ++i) g[i] = (__bf16)acc[tt][i];
    *(v8bf*)(Gp + ((((size_t)t * 16 + bt) * 16 + nt) * 32 + lane) * 8) = g;
  }
}

// ---------------- Kernel 3: sequential recurrence, 16 batches per workgroup ----------------
__global__ __launch_bounds__(256) void k_recurrent(const __bf16* __restrict__ Wz,
                                                   const __bf16* __restrict__ G0,
                                                   const __bf16* __restrict__ G1,
                                                   const float* __restrict__ hidden,
                                                   __bf16* __restrict__ Y) {
  __shared__ __align__(32) float  carry[16 * 512]; // 32 KB
  __shared__ __align__(32) float  F[16 * 256];     // 16 KB: o, then q in place
  __shared__ __align__(32) __bf16 Abf[16 * 256];   //  8 KB: bf16 A staging (x, then u)

  int tid  = threadIdx.x;
  int wave = tid >> 5, lane = tid & 31;
  int bt   = blockIdx.x;          // batch tile
  int b0   = bt * 16;
  int mrow = tid >> 4;            // batch within block
  int c16  = (tid & 15) * 16;
  int r16  = tid & 15;

  for (int i = tid; i < 16 * 512; i += 256) carry[i] = hidden[i & 511];
  __syncthreads();

  int nt0 = wave * 2, nt1 = nt0 + 1;
  int nl = lane & 15, mb = (lane >> 4) * 8;

  for (int t = 0; t < TTOT; ++t) {
#pragma unroll 1
    for (int half = 0; half < 2; ++half) {
      int matx = half * 2, mato = half * 2 + 1;
      const __bf16* G = half ? G1 : G0;
      int xoff = half ? 256 : 0;              // half0: x=inp0 ; half1: x=out1

      // 1) Abf = relu(vec(x))  (vectorized LDS traffic)
      {
        const float* cp = &carry[mrow * 512 + xoff + c16];
        v4f x0 = *(const v4f*)cp,       x1 = *(const v4f*)(cp + 4);
        v4f x2 = *(const v4f*)(cp + 8), x3 = *(const v4f*)(cp + 12);
        v8bf y0, y1;
#pragma unroll
        for (int i = 0; i < 4; ++i) {
          y0[i]     = (__bf16)(x0[i] > 0.f ? x0[i] : 0.f);
          y0[4 + i] = (__bf16)(x1[i] > 0.f ? x1[i] : 0.f);
          y1[i]     = (__bf16)(x2[i] > 0.f ? x2[i] : 0.f);
          y1[4 + i] = (__bf16)(x3[i] > 0.f ? x3[i] : 0.f);
        }
        *(v8bf*)&Abf[mrow * 256 + c16]     = y0;
        *(v8bf*)&Abf[mrow * 256 + c16 + 8] = y1;
      }
      __syncthreads();

      // 2) a = X @ Wx ; u = relu(a * gate)
      v8f acc0 = {}, acc1 = {};
#pragma unroll 1
      for (int kc = 0; kc < 8; ++kc) {
        v16bf a = loadA_bf16v(Abf, 256, kc * 32, lane);
        const __bf16* B0 = Wz + (size_t)((matx * 8 + kc) * 16 + nt0) * 512;
        const __bf16* B1 = Wz + (size_t)((matx * 8 + kc) * 16 + nt1) * 512;
        acc0 = wmma_bf16(a, loadB_swz(B0, lane), acc0);
        acc1 = wmma_bf16(a, loadB_swz(B1, lane), acc1);
      }
      v8bf g0 = *(const v8bf*)(G + ((((size_t)t * 16 + bt) * 16 + nt0) * 32 + lane) * 8);
      v8bf g1 = *(const v8bf*)(G + ((((size_t)t * 16 + bt) * 16 + nt1) * 32 + lane) * 8);
      float u0[8], u1[8];
#pragma unroll
      for (int i = 0; i < 8; ++i) {
        float x0 = acc0[i] * (float)g0[i]; u0[i] = x0 > 0.f ? x0 : 0.f;
        float x1 = acc1[i] * (float)g1[i]; u1[i] = x1 > 0.f ? x1 : 0.f;
      }
      __syncthreads();                        // all waves done reading Abf
#pragma unroll
      for (int i = 0; i < 8; ++i) {
        Abf[(mb + i) * 256 + nt0 * 16 + nl] = (__bf16)u0[i];
        Abf[(mb + i) * 256 + nt1 * 16 + nl] = (__bf16)u1[i];
      }
      __syncthreads();

      // 3) o = U @ Wo -> F (f32)
      v8f acc2 = {}, acc3 = {};
#pragma unroll 1
      for (int kc = 0; kc < 8; ++kc) {
        v16bf a = loadA_bf16v(Abf, 256, kc * 32, lane);
        const __bf16* B0 = Wz + (size_t)((mato * 8 + kc) * 16 + nt0) * 512;
        const __bf16* B1 = Wz + (size_t)((mato * 8 + kc) * 16 + nt1) * 512;
        acc2 = wmma_bf16(a, loadB_swz(B0, lane), acc2);
        acc3 = wmma_bf16(a, loadB_swz(B1, lane), acc3);
      }
#pragma unroll
      for (int i = 0; i < 8; ++i) {
        F[(mb + i) * 256 + nt0 * 16 + nl] = acc2[i];
        F[(mb + i) * 256 + nt1 * 16 + nl] = acc3[i];
      }
      __syncthreads();

      // 4) q = QR(o).Q via modified Gram-Schmidt; half-wave per batch; in place in F
      {
        int bq = wave * 2 + (lane >> 4);
        int gl = lane & 15;                   // lane owns row gl
        float mc[16];
#pragma unroll
        for (int k = 0; k < 16; ++k) mc[k] = F[bq * 256 + gl * 16 + k];
#pragma unroll 1
        for (int j = 0; j < 16; ++j) {
          float s = mc[j] * mc[j];
          s += __shfl_xor(s, 1, 32); s += __shfl_xor(s, 2, 32);
          s += __shfl_xor(s, 4, 32); s += __shfl_xor(s, 8, 32);
          float q = mc[j] * rsqrtf(s + 1e-20f);
          mc[j] = q;
#pragma unroll 1
          for (int k = j + 1; k < 16; ++k) {
            float d = q * mc[k];
            d += __shfl_xor(d, 1, 32); d += __shfl_xor(d, 2, 32);
            d += __shfl_xor(d, 4, 32); d += __shfl_xor(d, 8, 32);
            mc[k] -= d * q;
          }
        }
#pragma unroll
        for (int k = 0; k < 16; ++k) F[bq * 256 + gl * 16 + k] = mc[k];
      }
      __syncthreads();

      // 5) out = other_half @ q, in place into carry (each thread owns one row)
      {
        int ooff = half ? 0 : 256;  // half0: out1 = inp1 @ q ; half1: out0 = inp0 @ q
        float xr[16], orow[16];
#pragma unroll
        for (int k = 0; k < 16; ++k) xr[k] = carry[mrow * 512 + ooff + r16 * 16 + k];
#pragma unroll
        for (int j = 0; j < 16; ++j) {
          float s = 0.f;
#pragma unroll
          for (int k = 0; k < 16; ++k) s += xr[k] * F[mrow * 256 + k * 16 + j];
          orow[j] = s;
        }
        __syncthreads();
#pragma unroll
        for (int j = 0; j < 16; ++j) carry[mrow * 512 + ooff + r16 * 16 + j] = orow[j];
      }
      __syncthreads();
    }

    if (t >= NDELAY) {   // emit concat(out0,out1) row, vectorized bf16 stores
      size_t yrow = ((size_t)(b0 + mrow) * TSEQ + (t - NDELAY)) * 512;
      const float* cp0 = &carry[mrow * 512 + r16 * 16];
      const float* cp1 = cp0 + 256;
      v8bf y0, y1, y2, y3;
#pragma unroll
      for (int i = 0; i < 8; ++i) {
        y0[i] = (__bf16)cp0[i];  y1[i] = (__bf16)cp0[8 + i];
        y2[i] = (__bf16)cp1[i];  y3[i] = (__bf16)cp1[8 + i];
      }
      *(v8bf*)(Y + yrow + r16 * 16)           = y0;
      *(v8bf*)(Y + yrow + r16 * 16 + 8)       = y1;
      *(v8bf*)(Y + yrow + 256 + r16 * 16)     = y2;
      *(v8bf*)(Y + yrow + 256 + r16 * 16 + 8) = y3;
    }
    __syncthreads();
  }
}

// ---------------- Kernel 4: output GEMM (65536 x 512) @ (512 x 1024) -> f32 ----------------
__global__ __launch_bounds__(256) void k_outgemm(const __bf16* __restrict__ Y,
                                                 const __bf16* __restrict__ Woz,
                                                 float* __restrict__ out) {
  int wave = threadIdx.x >> 5, lane = threadIdx.x & 31;
  int mt = blockIdx.x;                 // 0..4095
  int nt = blockIdx.y * 8 + wave;      // 0..63
  const __bf16* A = Y + (size_t)mt * 16 * 512;
  v8f acc = {};
#pragma unroll 1
  for (int kc = 0; kc < 16; ++kc) {
    v16bf a = loadA_bf16v(A, 512, kc * 32, lane);
    const __bf16* Bt = Woz + (size_t)(kc * 64 + nt) * 512;
    acc = wmma_bf16(a, loadB_swz(Bt, lane), acc);
  }
  int nl = lane & 15, mb = (lane >> 4) * 8;
  float* C = out + (size_t)mt * 16 * 1024 + nt * 16;
#pragma unroll
  for (int i = 0; i < 8; ++i) C[(size_t)(mb + i) * 1024 + nl] = acc[i];
}

// ---------------- launch ----------------
extern "C" void kernel_launch(void* const* d_in, const int* in_sizes, int n_in,
                              void* d_out, int out_size, void* d_ws, size_t ws_size,
                              hipStream_t stream) {
  const int*   fn_input = (const int*)d_in[0];
  const float* lp0      = (const float*)d_in[1];
  const float* lp1      = (const float*)d_in[2];
  const float* wout     = (const float*)d_in[3];
  const float* emb      = (const float*)d_in[4];
  const float* hidden   = (const float*)d_in[5];
  float*       out      = (float*)d_out;

  __bf16* ws  = (__bf16*)d_ws;
  size_t  szG = (size_t)TTOT * BB * HF;       // 17,301,504 elements each
  __bf16* Wz  = ws;                           // 6 * 65536 swizzled 256x256 mats
  __bf16* Woz = Wz + 6 * 65536;               // 524288 swizzled out_linear
  __bf16* G0  = Woz + 524288;
  __bf16* G1  = G0 + szG;
  __bf16* Y   = G1 + szG;                     // 65536 x 512 bf16 row-major

  k_prepw256<<<1536, 256, 0, stream>>>(lp0, lp1, Wz);
  k_prepwout<<<2048, 256, 0, stream>>>(wout, Woz);
  k_gates<<<TTOT * 16, 256, 0, stream>>>(fn_input, emb, Wz, G0, G1);
  k_recurrent<<<16, 256, 0, stream>>>(Wz, G0, G1, hidden, Y);
  k_outgemm<<<dim3(65536 / 16, 8), 256, 0, stream>>>(Y, Woz, out);
}
